// model_LSTM_21689584845138
// MI455X (gfx1250) — matrix operand order repaired and checked by reading, not compiled, not run on hardware
//
#include <hip/hip_runtime.h>
#include <hip/hip_bf16.h>
#include <math.h>

// ---- model dims ----
#define B_   4096
#define T_   255
#define V_   32000
#define E_   64
#define H_   64
#define FH_  256
#define D1_  32
#define NT2_ 2000     // V_/16 column tiles of W2
#define NWXH_ (2 * 2 * 16 * 32 * 16)   // Wx+Wh fragment image: 32768 bf16 = 64KB

typedef __attribute__((ext_vector_type(16))) __bf16 v16bf;
typedef __attribute__((ext_vector_type(8)))  float  v8f;

struct U32B { uint4 a, b; };   // 32 bytes == one v16bf fragment

__device__ __forceinline__ unsigned short f2bfu(float f) {
  __bf16 h = (__bf16)f;                       // native v_cvt, RNE
  return __builtin_bit_cast(unsigned short, h);
}
__device__ __forceinline__ float sigm(float x) { return 1.0f / (1.0f + __expf(-x)); }
__device__ __forceinline__ float tanh_(float x) {
  float e = __expf(2.0f * x);
  return (e - 1.0f) / (e + 1.0f);
}
// load a 32-byte fragment from two 16-byte chunks
__device__ __forceinline__ v16bf ld_frag2(const unsigned short* p0, const unsigned short* p1) {
  U32B u; u.a = *(const uint4*)p0; u.b = *(const uint4*)p1;
  return __builtin_bit_cast(v16bf, u);
}

// ===================== kernel 0a: swizzle W2 [32][32000] -> bf16 B-fragment layout ================
// wsW2[nt][lane][half] ; B(k,n): lane = n%16 + 16*(k>=16), half = k%16
__global__ __launch_bounds__(256) void swz_w2_kernel(const float* __restrict__ W2,
                                                     unsigned short* __restrict__ wsW2) {
  int i = blockIdx.x * blockDim.x + threadIdx.x;
  if (i >= NT2_ * 32 * 16) return;
  int half = i & 15, ln = (i >> 4) & 31, nt = i >> 9;
  int k = half + 16 * (ln >> 4);
  int n = nt * 16 + (ln & 15);
  wsW2[i] = f2bfu(W2[k * V_ + n]);
}

// ===================== kernel 0b: embeddings f32 -> bf16, pre-swizzled in A-fragment order =======
// Row of 64 reordered into 4 chunks of 16: chunk c = 2*kt + laneHi holds
// k = kt*32 + laneHi*8 + {0..7, +16..+23}. A lane then reads one contiguous 32B span per k-tile.
__global__ __launch_bounds__(256) void cvt_emb_kernel(const float* __restrict__ emb,
                                                      unsigned short* __restrict__ wsEmb) {
  int i = blockIdx.x * blockDim.x + threadIdx.x;
  if (i >= V_ * E_) return;
  int row = i >> 6, pos = i & 63;
  int chunk = pos >> 4, within = pos & 15;
  int kt = chunk >> 1, hi8 = chunk & 1;
  int k = kt * 32 + hi8 * 8 + (within & 7) + 16 * (within >> 3);
  wsEmb[i] = f2bfu(emb[(size_t)row * E_ + k]);
}

// ===================== kernel 0c: Wx/Wh -> bf16 B-fragment image =====================
// wsWxh[src2][kt2][nt16][lane32][half16] ; element (k,n): lane = n%16 + 16*(k%16>=8), half = k%8 + 8*(k%32>=16)
__global__ __launch_bounds__(256) void swz_wxh_kernel(const float* __restrict__ Wx,
                                                      const float* __restrict__ Wh,
                                                      unsigned short* __restrict__ wsWxh) {
  int i = blockIdx.x * blockDim.x + threadIdx.x;
  if (i >= NWXH_) return;
  int half = i & 15, ln = (i >> 4) & 31, nt = (i >> 9) & 15, kt = (i >> 13) & 1, src = (i >> 14) & 1;
  int k = kt * 32 + (half & 7) + 16 * (half >> 3) + 8 * (ln >> 4);
  int n = nt * 16 + (ln & 15);
  const float* W = src ? Wh : Wx;
  wsWxh[i] = f2bfu(W[k * FH_ + n]);
}

// ===================== kernel 1: LSTM over T steps (1 wave / workgroup, 256 workgroups) ==========
__global__ __launch_bounds__(32) void lstm_kernel(
    const int* __restrict__ idxs, const unsigned short* __restrict__ wsEmb,
    const unsigned short* __restrict__ wsWxh,
    const float* __restrict__ bl, float* __restrict__ wsH) {
  extern __shared__ unsigned short smem[];
  unsigned short* ldsW = smem;            // [src2][kt2][nt16][lane32][half16] = 32768 ush (64KB)
  unsigned short* ldsH = smem + NWXH_;    // [16][64] = 1024 ush (2KB)

  const int lane = threadIdx.x & 31;
  const int lanelo = lane & 15, laneHi = lane >> 4;

  // stage the pre-swizzled weight image: pure 64KB vector copy (global b128 -> ds b128)
  {
    const uint4* s4 = (const uint4*)wsWxh;
    uint4* d4 = (uint4*)ldsW;
    for (int i = lane; i < NWXH_ / 8; i += 32) d4[i] = s4[i];
    uint4 z; z.x = z.y = z.z = z.w = 0u;
    uint4* h4 = (uint4*)ldsH;
    for (int i = lane; i < (16 * 64) / 8; i += 32) h4[i] = z;   // h0 = 0
  }
  __syncthreads();   // single-wave workgroup: ~free (S_NOP), acts as LDS scheduling fence

  const int b0 = blockIdx.x * 16;        // this wave's batch rows
  const int mrow = lanelo;               // A-fragment row per lane
  const int koff = 8 * laneHi;           // A-layout K interleave

  float bias[16];
#pragma unroll
  for (int nt = 0; nt < 16; ++nt) bias[nt] = bl[nt * 16 + lanelo];

  v8f c[4];
#pragma unroll
  for (int j = 0; j < 4; ++j)
#pragma unroll
    for (int r = 0; r < 8; ++r) c[j][r] = 0.0f;

  const unsigned short* hbase = ldsH + mrow * 64;
  const int* myIdx = idxs + (size_t)(b0 + mrow) * T_;

  for (int t = 0; t < T_; ++t) {
    // ---- A-fragments: x_t gathered straight as pre-swizzled bf16 (no conversion VALU)
    const unsigned short* rowp = wsEmb + (size_t)myIdx[t] * E_ + laneHi * 16;
    // prefetch next step's row while this step's WMMAs run
    if (t + 1 < T_) {
      const unsigned short* nrow = wsEmb + (size_t)myIdx[t + 1] * E_ + laneHi * 16;
      __builtin_prefetch(nrow, 0, 0);
    }
    v16bf ax[2], ah[2];
#pragma unroll
    for (int kt = 0; kt < 2; ++kt) {
      ax[kt] = ld_frag2(rowp + kt * 32, rowp + kt * 32 + 8);
      int base = kt * 32 + koff;
      ah[kt] = ld_frag2(hbase + base, hbase + base + 16);
    }

    // ---- z = x@Wx + h@Wh + b, gate-interleaved: process sub-column j with its 4 gate tiles
#pragma unroll
    for (int j = 0; j < 4; ++j) {
      v8f zt[4];
#pragma unroll
      for (int g = 0; g < 4; ++g) {
        const int nt = g * 4 + j;
        v8f acc;
#pragma unroll
        for (int r = 0; r < 8; ++r) acc[r] = bias[nt];
        const unsigned short* p;
        p = ldsW + ((size_t)(0 * 16 + nt) * 32 + lane) * 16;   // Wx, kt=0
        acc = __builtin_amdgcn_wmma_f32_16x16x32_bf16(false, ax[0], false, ld_frag2(p, p + 8),
                                                      (short)0, acc, false, false);
        p = ldsW + ((size_t)(1 * 16 + nt) * 32 + lane) * 16;   // Wx, kt=1
        acc = __builtin_amdgcn_wmma_f32_16x16x32_bf16(false, ax[1], false, ld_frag2(p, p + 8),
                                                      (short)0, acc, false, false);
        p = ldsW + ((size_t)(2 * 16 + nt) * 32 + lane) * 16;   // Wh, kt=0
        acc = __builtin_amdgcn_wmma_f32_16x16x32_bf16(false, ah[0], false, ld_frag2(p, p + 8),
                                                      (short)0, acc, false, false);
        p = ldsW + ((size_t)(3 * 16 + nt) * 32 + lane) * 16;   // Wh, kt=1
        acc = __builtin_amdgcn_wmma_f32_16x16x32_bf16(false, ah[1], false, ld_frag2(p, p + 8),
                                                      (short)0, acc, false, false);
        zt[g] = acc;
      }
      // gates in fp32 (D-layout: row = r + 8*laneHi, col = j*16 + lanelo)
#pragma unroll
      for (int r = 0; r < 8; ++r) {
        float iv = sigm(zt[0][r]);
        float fv = sigm(zt[1][r]);
        float gv = tanh_(zt[2][r]);
        float ov = sigm(zt[3][r]);
        float cn = fv * c[j][r] + iv * gv;
        c[j][r] = cn;
        float hv = ov * tanh_(cn);
        ldsH[(r + 8 * laneHi) * 64 + j * 16 + lanelo] = f2bfu(hv);
        if (t == T_ - 1)
          wsH[(size_t)(b0 + r + 8 * laneHi) * H_ + j * 16 + lanelo] = hv;
      }
    }
    __syncthreads();   // free in 1-wave WG; keeps LDS handoff ordered for next step
  }
}

// ===================== kernel 2: out = relu(h @ W1 + b1) =====================
__global__ __launch_bounds__(256) void head1_kernel(const float* __restrict__ wsH,
                                                    const float* __restrict__ W1,
                                                    const float* __restrict__ b1,
                                                    float* __restrict__ wsOut) {
  int i = blockIdx.x * blockDim.x + threadIdx.x;
  if (i >= B_ * D1_) return;
  int bb = i >> 5, d = i & 31;
  float s = b1[d];
  const float* hrow = wsH + (size_t)bb * H_;
#pragma unroll 8
  for (int e = 0; e < H_; ++e) s = fmaf(hrow[e], W1[e * D1_ + d], s);
  wsOut[i] = fmaxf(s, 0.0f);
}

// ===================== kernel 3: logits + online softmax (single HBM write of output) ============
#define SWAVES 4
__global__ __launch_bounds__(128) void head2_kernel(const float* __restrict__ wsOut,
                                                    const unsigned short* __restrict__ wsW2,
                                                    const float* __restrict__ b2,
                                                    float* __restrict__ out) {
  __shared__ float ldsM[SWAVES][16], ldsS[SWAVES][16];
  const int tid = threadIdx.x, wave = tid >> 5, lane = tid & 31;
  const int lanelo = lane & 15, laneHi = lane >> 4;
  const int b0 = blockIdx.x * 16;

  // A-fragment of relu-out tile [16 x 32] (K = D1 = 32 exactly -> one WMMA per n-tile)
  v16bf a;
#pragma unroll
  for (int hi = 0; hi < 16; ++hi) {
    int k = hi + 16 * laneHi;
    a[hi] = (__bf16)wsOut[(size_t)(b0 + lanelo) * D1_ + k];
  }

  float m[8], s[8];
#pragma unroll
  for (int r = 0; r < 8; ++r) { m[r] = -1e30f; s[r] = 0.0f; }

  // ---- pass 1: online max/sum over this wave's n-tiles
  for (int nt = wave; nt < NT2_; nt += SWAVES) {
    const unsigned short* p = wsW2 + ((size_t)nt * 32 + lane) * 16;
    v16bf bf = ld_frag2(p, p + 8);
    float bv = b2[nt * 16 + lanelo];
    v8f acc;
#pragma unroll
    for (int r = 0; r < 8; ++r) acc[r] = bv;
    acc = __builtin_amdgcn_wmma_f32_16x16x32_bf16(false, a, false, bf, (short)0, acc, false, false);
#pragma unroll
    for (int r = 0; r < 8; ++r) {
      float x = acc[r];
      float mn = fmaxf(m[r], x);
      s[r] = s[r] * __expf(m[r] - mn) + __expf(x - mn);
      m[r] = mn;
    }
  }
  // combine the 16 lanes sharing the same rows (xor within each 16-lane half)
#pragma unroll
  for (int mask = 1; mask <= 8; mask <<= 1) {
#pragma unroll
    for (int r = 0; r < 8; ++r) {
      float om = __shfl_xor(m[r], mask, 32);
      float os = __shfl_xor(s[r], mask, 32);
      float mn = fmaxf(m[r], om);
      s[r] = s[r] * __expf(m[r] - mn) + os * __expf(om - mn);
      m[r] = mn;
    }
  }
  if (lanelo == 0) {
#pragma unroll
    for (int r = 0; r < 8; ++r) { ldsM[wave][r + 8 * laneHi] = m[r]; ldsS[wave][r + 8 * laneHi] = s[r]; }
  }
  __syncthreads();
  float M[8], RS[8];
#pragma unroll
  for (int r = 0; r < 8; ++r) {
    int rowi = r + 8 * laneHi;
    float mm = -1e30f, ss = 0.0f;
#pragma unroll
    for (int w = 0; w < SWAVES; ++w) {
      float om = ldsM[w][rowi], os = ldsS[w][rowi];
      float mn = fmaxf(mm, om);
      ss = ss * __expf(mm - mn) + os * __expf(om - mn);
      mm = mn;
    }
    M[r] = mm; RS[r] = 1.0f / ss;
  }

  // ---- pass 2: recompute logits (cheap, L2-resident) and write normalized probs once
  for (int nt = wave; nt < NT2_; nt += SWAVES) {
    const unsigned short* p = wsW2 + ((size_t)nt * 32 + lane) * 16;
    v16bf bf = ld_frag2(p, p + 8);
    float bv = b2[nt * 16 + lanelo];
    v8f acc;
#pragma unroll
    for (int r = 0; r < 8; ++r) acc[r] = bv;
    acc = __builtin_amdgcn_wmma_f32_16x16x32_bf16(false, a, false, bf, (short)0, acc, false, false);
#pragma unroll
    for (int r = 0; r < 8; ++r) {
      out[(size_t)(b0 + r + 8 * laneHi) * V_ + nt * 16 + lanelo] = __expf(acc[r] - M[r]) * RS[r];
    }
  }
}

// ===================== launch =====================
extern "C" void kernel_launch(void* const* d_in, const int* in_sizes, int n_in,
                              void* d_out, int out_size, void* d_ws, size_t ws_size,
                              hipStream_t stream) {
  const int*   idxs = (const int*)d_in[0];
  const float* emb  = (const float*)d_in[1];
  const float* Wx   = (const float*)d_in[2];
  const float* Wh   = (const float*)d_in[3];
  const float* bl   = (const float*)d_in[4];
  const float* W1   = (const float*)d_in[5];
  const float* b1   = (const float*)d_in[6];
  const float* W2   = (const float*)d_in[7];
  const float* b2   = (const float*)d_in[8];
  float* out = (float*)d_out;

  // workspace layout (~7.6 MB)
  float* wsH   = (float*)d_ws;                                         // [4096][64] f32
  float* wsOut = wsH + (size_t)B_ * H_;                                // [4096][32] f32
  unsigned short* wsW2  = (unsigned short*)(wsOut + (size_t)B_ * D1_); // [2000][32][16] bf16
  unsigned short* wsEmb = wsW2 + (size_t)NT2_ * 32 * 16;               // [32000][64] bf16 (swizzled)
  unsigned short* wsWxh = wsEmb + (size_t)V_ * E_;                     // [32768] bf16 fragment image

  swz_w2_kernel<<<(NT2_ * 32 * 16 + 255) / 256, 256, 0, stream>>>(W2, wsW2);
  cvt_emb_kernel<<<(V_ * E_ + 255) / 256, 256, 0, stream>>>(emb, wsEmb);
  swz_wxh_kernel<<<(NWXH_ + 255) / 256, 256, 0, stream>>>(Wx, Wh, wsWxh);

  size_t shmem = (size_t)(NWXH_ + 16 * 64) * sizeof(unsigned short);   // 66KB
  lstm_kernel<<<B_ / 16, 32, shmem, stream>>>(idxs, wsEmb, wsWxh, bl, wsH);

  head1_kernel<<<(B_ * D1_ + 255) / 256, 256, 0, stream>>>(wsH, W1, b1, wsOut);

  head2_kernel<<<B_ / 16, 128, 0, stream>>>(wsOut, wsW2, b2, out);
}
